// WeightedLossModel_13065290515034
// MI455X (gfx1250) — matrix operand order repaired and checked by reading, not compiled, not run on hardware
//
#include <hip/hip_runtime.h>
#include <hip/hip_bf16.h>
#include <math.h>
#include <stdint.h>

typedef __attribute__((ext_vector_type(16))) __bf16 v16bf;
typedef __attribute__((ext_vector_type(8)))  float  v8f;
typedef __attribute__((ext_vector_type(4)))  unsigned int u32x4;
typedef __attribute__((ext_vector_type(8)))  int i32x8;
typedef __attribute__((ext_vector_type(4)))  int i32x4;

#define NB   512
#define HH   96
#define WW   96
#define PIX  (HH * WW)
#define KH   31

__device__ __forceinline__ float clipf(float x, float lo, float hi) {
    return fminf(fmaxf(x, lo), hi);
}

__device__ __forceinline__ float warpSum(float v) {
#pragma unroll
    for (int o = 16; o > 0; o >>= 1) v += __shfl_down(v, o, 32);
    return v;
}

// ---------------------------------------------------------------------------
// Tensor Data Mover: DMA a contiguous run of f32 from global into LDS.
// 1-D descriptor per CDNA5 ISA ch.8 (6-arg builtin form on this toolchain).
// ---------------------------------------------------------------------------
__device__ __forceinline__ void tdm_load_f32(void* lds_dst, const float* src,
                                             unsigned nelem) {
    unsigned long long ga = (unsigned long long)(uintptr_t)src;
    unsigned ga_lo   = __builtin_amdgcn_readfirstlane((unsigned)ga);
    unsigned ga_hi   = __builtin_amdgcn_readfirstlane((unsigned)(ga >> 32));
    unsigned lds_off = __builtin_amdgcn_readfirstlane((unsigned)(uintptr_t)lds_dst);
    u32x4 g0;
    g0[0] = 1u;                                   // count=1, user descriptor
    g0[1] = lds_off;                              // lds_addr (bytes)
    g0[2] = ga_lo;                                // global_addr[31:0]
    g0[3] = (ga_hi & 0x01ffffffu) | (2u << 30);   // global_addr[56:32], type=2
    i32x8 g1;
    g1[0] = (int)(2u << 16);                          // data_size = 4 bytes
    g1[1] = (int)((nelem & 0xffffu) << 16);           // tensor_dim0[15:0]
    g1[2] = (int)((nelem >> 16) & 0xffffu) | (1 << 16); // dim0[31:16], dim1=1
    g1[3] = (int)((nelem & 0xffffu) << 16);           // tile_dim0 = nelem
    g1[4] = 0;                                        // tile_dim1/2 unused
    g1[5] = (int)nelem;                               // tensor_dim0_stride
    g1[6] = 0;
    g1[7] = 0;
    i32x4 z4 = {0, 0, 0, 0};
    i32x8 z8 = {0, 0, 0, 0, 0, 0, 0, 0};
    __builtin_amdgcn_tensor_load_to_lds(g0, g1, z4, z4, z8, 0);
}

// ---------------------------------------------------------------------------
// Kernel 1: weighted MSE partial sum (bandwidth bound, float4 traffic)
// ---------------------------------------------------------------------------
__global__ void __launch_bounds__(256) mse_kernel(const float4* __restrict__ yp,
                                                  const float4* __restrict__ yt,
                                                  const float4* __restrict__ wm,
                                                  float* __restrict__ ws, int n4) {
    __shared__ float sbuf[8];
    float acc = 0.f;
    for (int i = blockIdx.x * blockDim.x + threadIdx.x; i < n4;
         i += gridDim.x * blockDim.x) {
        float4 a = yp[i], b = yt[i], w = wm[i];
        float d0 = a.x - b.x, d1 = a.y - b.y, d2 = a.z - b.z, d3 = a.w - b.w;
        acc += d0 * d0 * w.x + d1 * d1 * w.y + d2 * d2 * w.z + d3 * d3 * w.w;
    }
    acc = warpSum(acc);
    int lane = threadIdx.x & 31, wv = threadIdx.x >> 5;
    if (lane == 0) sbuf[wv] = acc;
    __syncthreads();
    if (threadIdx.x == 0) {
        float s = 0.f;
        for (int k = 0; k < 8; k++) s += sbuf[k];
        atomicAdd(&ws[0], s);
    }
}

// ---------------------------------------------------------------------------
// Kernel 2: adaptive-moment ellipticity. One WG per image; image DMA'd into
// LDS once by the TDM, then 40 fixed iterations entirely on-chip.
// ---------------------------------------------------------------------------
__global__ void __launch_bounds__(256) ellip_kernel(const float* __restrict__ yt,
                                                    const float* __restrict__ yp,
                                                    float* __restrict__ ws) {
    int src = (blockIdx.x >= NB) ? 1 : 0;
    int b   = blockIdx.x - src * NB;
    const float* img = (src ? yp : yt) + (size_t)b * PIX;

    __shared__ float sImg[PIX];    // 36 KB, resident for all 40 iterations
    __shared__ float sred[8 * 8];  // per-wave partials: [wave][0..6]
    __shared__ float sbuf[8];
    __shared__ float st[9];        // ax,ay,axy,mux,muy,back,e1,e2,total

    int lane = threadIdx.x & 31, wv = threadIdx.x >> 5;

    if (wv == 0) {
        tdm_load_f32(sImg, img, PIX);
        __builtin_amdgcn_s_wait_tensorcnt(0);
    }
    __syncthreads();

    float tot = 0.f;
    for (int i = threadIdx.x; i < PIX; i += 256) tot += sImg[i];
    tot = warpSum(tot);
    if (lane == 0) sbuf[wv] = tot;
    __syncthreads();
    if (threadIdx.x == 0) {
        float s = 0.f;
        for (int k = 0; k < 8; k++) s += sbuf[k];
        st[0] = 2.f; st[1] = 2.f; st[2] = 0.f; st[3] = 0.f; st[4] = 0.f;
        st[5] = 0.f; st[6] = 0.f; st[7] = 0.f; st[8] = s;
    }
    __syncthreads();

    int hInit = threadIdx.x / 96;                 // 0..2
    int wInit = threadIdx.x - hInit * 96;

    for (int it = 0; it < 40; it++) {
        float ax  = clipf(st[0], 0.5f, 20.f);
        float ay  = clipf(st[1], 0.5f, 20.f);
        float axy = clipf(st[2], -10.f, 10.f);
        float mux = st[3], muy = st[4], back = st[5];

        float ad  = fmaxf(ax * ay, 1e-8f);
        float ar  = clipf(axy / ad, -0.95f, 0.95f);
        float arb = fmaxf(2.f * (1.f - ar * ar), 1e-8f);
        float sq  = fmaxf(1.f - ar * ar, 1e-8f);
        float A   = 1.f / (6.283185307179586f * ad * sqrtf(sq));
        A = clipf(A, 1e-10f, 1e10f);
        float axsq = fmaxf(ax * ax, 1e-8f);
        float aysq = fmaxf(ay * ay, 1e-8f);
        float i1 = 1.f / (arb * axsq);
        float i2 = 1.f / (arb * aysq);
        float c3 = 2.f * axy / (arb * axsq * aysq);

        float t0 = 0, t1 = 0, t2 = 0, t3 = 0, t4 = 0, t5 = 0, t6 = 0;
        int h = hInit, wx = wInit;
        int idx = threadIdx.x;
#pragma unroll 4
        for (int rep = 0; rep < 36; rep++) {      // 9216 / 256 == 36 exactly
            float X  = (float)wx - 47.5f;
            float Y  = (float)h  - 47.5f;
            float Xc = X - mux, Yc = Y - muy;
            float et = Xc * Xc * i1 + Yc * Yc * i2 - c3 * Xc * Yc;
            et = clipf(et, 0.f, 30.f);
            float k  = A * __expf(-et);
            float ik = (sImg[idx] - back) * k;
            t0 += X * Y * ik; t1 += ik;         t2 += X * ik; t3 += Y * ik;
            t4 += X * X * ik; t5 += Y * Y * ik; t6 += k * k;
            idx += 256; wx += 64; h += 2;
            if (wx >= 96) { wx -= 96; h += 1; }
        }
        float tv[7] = {t0, t1, t2, t3, t4, t5, t6};
#pragma unroll
        for (int s = 0; s < 7; s++) {
            float v = warpSum(tv[s]);
            if (lane == 0) sred[wv * 8 + s] = v;
        }
        __syncthreads();
        if (threadIdx.x == 0) {
            float T[7];
#pragma unroll
            for (int s = 0; s < 7; s++) {
                float r = 0.f;
                for (int k = 0; k < 8; k++) r += sred[k * 8 + s];
                T[s] = r;
            }
            float t2s  = fmaxf(fabsf(T[1]), 1e-6f);
            float t7s  = fmaxf(T[6], 1e-6f);
            float flux = T[1] / t7s;
            float nb   = clipf((st[8] - flux) / (float)PIX, -1.f, 1.f);
            float nmux = clipf(T[2] / t2s, -48.f, 48.f);
            float nmuy = clipf(T[3] / t2s, -48.f, 48.f);
            float sxx  = T[4] / t2s - (T[2] / t2s) * (T[2] / t2s);
            float syy  = T[5] / t2s - (T[3] / t2s) * (T[3] / t2s);
            float sxy  = T[0] / t2s - T[2] * T[3] / (t2s * t2s);
            sxx = clipf(sxx, 0.25f, 100.f);
            syy = clipf(syy, 0.25f, 100.f);
            sxy = clipf(sxy, -50.f, 50.f);
            float den = fmaxf(sxx + syy, 1e-6f);
            st[0] = sqrtf(clipf(sxx * 2.f, 1.f, 400.f));
            st[1] = sqrtf(clipf(syy * 2.f, 1.f, 400.f));
            st[2] = clipf(2.f * sxy, -20.f, 20.f);
            st[3] = nmux; st[4] = nmuy; st[5] = nb;
            st[6] = clipf((sxx - syy) / den, -0.95f, 0.95f);
            st[7] = clipf(2.f * sxy / den, -0.95f, 0.95f);
        }
        __syncthreads();
    }
    if (threadIdx.x == 0) {
        ws[3 + src * 1024 + b]       = st[6];
        ws[3 + src * 1024 + 512 + b] = st[7];
    }
}

// ---------------------------------------------------------------------------
// Kernel 3: per-sample 31x31 SAME conv via bf16-split WMMA (3xBF16 trick).
// The image is pre-split ONCE into hi/lo bf16 planes with 16 zero-pad columns
// each side (px = x+16), so the steady-state A-build is pure 16-bit LDS
// gathers: no conversions, no clamping, no edge branches.
//   A[m,k] = Img[r][x0+m+k-15]  (Toeplitz over a row, K=32 padded taps)
//   B[k,n] = Wk[n + 16*blk][k]  (transposed normalized kernel, 2 col-blocks)
//   D[m,n] -> Out[r+15-(n+16*blk)][x0+m]
// Guard rows make the ds_add_f32 scatter branch-free; row stride 17 makes the
// 16 row-strided lane addresses hit 16 distinct LDS banks.
// ---------------------------------------------------------------------------
#define OSTR 17
#define OROWS 128            // y+16 in [0,127] for y in [-16,110]
#define PSTR 128             // padded bf16 row stride: px = x+16, x in [-16,111]

__global__ void __launch_bounds__(256) reblur_kernel(const float* __restrict__ yp,
                                                     const float* __restrict__ blur,
                                                     const float* __restrict__ wm,
                                                     const float* __restrict__ ker,
                                                     float* __restrict__ ws) {
    int b = blockIdx.x;
    __shared__ __bf16 sH[HH * PSTR];        // 24 KB hi-plane, zero-padded
    __shared__ __bf16 sL[HH * PSTR];        // 24 KB lo-plane, zero-padded
    __shared__ float  sKer[32 * 32];        // 4 KB, zero-padded to 32x32
    __shared__ float  sOut[OROWS * OSTR];   // 8.5 KB guarded accumulator
    __shared__ float  sbuf[8];
    __shared__ float  sKinv;

    const float* img = yp  + (size_t)b * PIX;
    const float* kr  = ker + (size_t)b * KH * KH;
    int lane = threadIdx.x & 31, wv = threadIdx.x >> 5;

    // one-time hi/lo bf16 split into padded planes (zero pad = SAME padding)
    for (int i = threadIdx.x; i < HH * PSTR; i += 256) {
        int row = i >> 7, px = i & 127, x = px - 16;
        float v = (x >= 0 && x < WW) ? img[row * WW + x] : 0.f;
        __bf16 h = (__bf16)v;
        sH[i] = h;
        sL[i] = (__bf16)(v - (float)h);
    }

    float ksum = 0.f;
    for (int i = threadIdx.x; i < 1024; i += 256) {
        int ki = i >> 5, kj = i & 31;
        float v = (ki < KH && kj < KH) ? kr[ki * KH + kj] : 0.f;
        sKer[i] = v;
        ksum += v;
    }
    ksum = warpSum(ksum);
    if (lane == 0) sbuf[wv] = ksum;
    __syncthreads();
    if (threadIdx.x == 0) {
        float s = 0.f;
        for (int k = 0; k < 8; k++) s += sbuf[k];
        sKinv = 1.f / (s + 1e-6f);
    }
    __syncthreads();
    float kinv = sKinv;
    for (int i = threadIdx.x; i < 1024; i += 256) sKer[i] *= kinv;
    __syncthreads();                        // planes + normalized sKer ready

    bool hi = lane >= 16;
    int  nn = lane & 15;

    // B matrices (loop-invariant): 2 column-blocks x (hi,lo) bf16 split
    v16bf b0h, b0l, b1h, b1l;
#pragma unroll
    for (int e = 0; e < 16; e++) {
        int j = e + ((e >= 8) ? 8 : 0) + (hi ? 8 : 0); // K index held by this lane
        float w0 = sKer[nn * 32 + j];              // kernel row i = nn
        float w1 = sKer[(16 + nn) * 32 + j];       // kernel row i = 16+nn
        __bf16 h0 = (__bf16)w0; b0h[e] = h0; b0l[e] = (__bf16)(w0 - (float)h0);
        __bf16 h1 = (__bf16)w1; b1h[e] = h1; b1l[e] = (__bf16)(w1 - (float)h1);
    }

    float lacc = 0.f;
    for (int xt = 0; xt < 6; xt++) {
        int x0 = xt * 16;
        for (int i = threadIdx.x; i < OROWS * OSTR; i += 256) sOut[i] = 0.f;
        __syncthreads();

        // per-lane base: px = x0 + nn + j - 15 + 16, j = (hi?8:0) + off(e)
        int pbase = x0 + nn + 1 + (hi ? 8 : 0);

        for (int r = wv; r < 96; r += 8) {         // uniform 12 trips per wave
            const __bf16* rH = &sH[r * PSTR + pbase];
            const __bf16* rL = &sL[r * PSTR + pbase];
            v16bf ah, al;
#pragma unroll
            for (int e = 0; e < 16; e++) {
                int off = e + ((e >= 8) ? 8 : 0);  // taps {0..7} u {16..23}
                ah[e] = rH[off];
                al[e] = rL[off];
            }
            v8f c0 = {}; v8f c1 = {};
            c0 = __builtin_amdgcn_wmma_f32_16x16x32_bf16(false, al, false, b0h, (short)0, c0, false, false);
            c1 = __builtin_amdgcn_wmma_f32_16x16x32_bf16(false, al, false, b1h, (short)0, c1, false, false);
            c0 = __builtin_amdgcn_wmma_f32_16x16x32_bf16(false, ah, false, b0l, (short)0, c0, false, false);
            c1 = __builtin_amdgcn_wmma_f32_16x16x32_bf16(false, ah, false, b1l, (short)0, c1, false, false);
            c0 = __builtin_amdgcn_wmma_f32_16x16x32_bf16(false, ah, false, b0h, (short)0, c0, false, false);
            c1 = __builtin_amdgcn_wmma_f32_16x16x32_bf16(false, ah, false, b1h, (short)0, c1, false, false);

            // Branch-free scatter: row is per-lane constant across all 8 elems.
            int base0 = (r + 31 - nn) * OSTR + (hi ? 8 : 0);  // y0+16
            int base1 = (r + 15 - nn) * OSTR + (hi ? 8 : 0);  // y1+16
#pragma unroll
            for (int v = 0; v < 8; v++) {
                atomicAdd(&sOut[base0 + v], c0[v]);
                atomicAdd(&sOut[base1 + v], c1[v]);
            }
        }
        __syncthreads();

        for (int i = threadIdx.x; i < 96 * 16; i += 256) {
            int y = i >> 4, xm = i & 15, x = x0 + xm;
            float o  = sOut[(y + 16) * OSTR + xm];
            float bl = blur[(((size_t)b * 96 + y) * 96 + x) * 2];  // channel 0
            float w  = wm[((size_t)b * 96 + y) * 96 + x];
            float d  = o - bl;
            lacc += d * d * w;
        }
        __syncthreads();
    }

    lacc = warpSum(lacc);
    if (lane == 0) sbuf[wv] = lacc;
    __syncthreads();
    if (threadIdx.x == 0) {
        float s = 0.f;
        for (int k = 0; k < 8; k++) s += sbuf[k];
        atomicAdd(&ws[1], s);
    }
}

// ---------------------------------------------------------------------------
// Kernel 4: ellipticity difference reduction over 512 samples.
// ---------------------------------------------------------------------------
__global__ void __launch_bounds__(256) ellip_diff_kernel(float* __restrict__ ws) {
    __shared__ float sbuf[8];
    float acc = 0.f;
    for (int i = threadIdx.x; i < NB; i += 256) {
        float d1 = ws[3 + i]       - ws[3 + 1024 + i];
        float d2 = ws[3 + 512 + i] - ws[3 + 1024 + 512 + i];
        acc += d1 * d1 + d2 * d2;
    }
    acc = warpSum(acc);
    int lane = threadIdx.x & 31, wv = threadIdx.x >> 5;
    if (lane == 0) sbuf[wv] = acc;
    __syncthreads();
    if (threadIdx.x == 0) {
        float s = 0.f;
        for (int k = 0; k < 8; k++) s += sbuf[k];
        ws[2] = s;
    }
}

// ---------------------------------------------------------------------------
// Kernel 5: combine losses.
// ---------------------------------------------------------------------------
__global__ void finalize_kernel(const float* __restrict__ ws,
                                const int* __restrict__ epoch,
                                float* __restrict__ out) {
    float inv = 1.f / (float)((size_t)NB * PIX);
    float mse = ws[0] * inv;
    float reb = ws[1] * inv;
    float ed  = clipf(ws[2] / (float)NB, 0.f, 1.f);
    int e = epoch[0];
    if (e > 4) e = 4;
    float wgt = 0.01f + 0.1f * (float)e;
    out[0] = 100.f * mse + 100.f * reb + wgt * ed;
}

// ---------------------------------------------------------------------------
extern "C" void kernel_launch(void* const* d_in, const int* in_sizes, int n_in,
                              void* d_out, int out_size, void* d_ws, size_t ws_size,
                              hipStream_t stream) {
    const float* yp    = (const float*)d_in[0];   // y_pred   [512,96,96,1]
    const float* yt    = (const float*)d_in[1];   // y_true   [512,96,96,1]
    const float* blur  = (const float*)d_in[2];   // blurred  [512,96,96,2]
    const float* ker   = (const float*)d_in[3];   // kernel   [512,31,31,1]
    const float* wm    = (const float*)d_in[4];   // weights  [512,96,96,1]
    const int*   epoch = (const int*)d_in[5];
    float* ws  = (float*)d_ws;
    float* out = (float*)d_out;

    (void)hipMemsetAsync(d_ws, 0, 16, stream);    // zero atomic slots

    int n4 = (NB * PIX) / 4;
    mse_kernel<<<2048, 256, 0, stream>>>((const float4*)yp, (const float4*)yt,
                                         (const float4*)wm, ws, n4);
    ellip_kernel<<<2 * NB, 256, 0, stream>>>(yt, yp, ws);
    reblur_kernel<<<NB, 256, 0, stream>>>(yp, blur, wm, ker, ws);
    ellip_diff_kernel<<<1, 256, 0, stream>>>(ws);
    finalize_kernel<<<1, 1, 0, stream>>>(ws, epoch, out);
}
